// BiLSTM_77034533421798
// MI455X (gfx1250) — matrix operand order repaired and checked by reading, not compiled, not run on hardware
//
#include <hip/hip_runtime.h>
#include <hip/hip_bf16.h>
#include <math.h>

// ---------------------------------------------------------------------------
// BiLSTM for MI455X (gfx1250): persistent WMMA kernel, bf16 matmul, fp32 acc.
// Weights staged in LDS (320KB/WGP); A-fragments software-pipelined; cell
// state held in registers. T=512, B=128, D_IN=H=400, L=2, bidirectional.
// ---------------------------------------------------------------------------

typedef __bf16 bf16_t;
typedef bf16_t v16bf __attribute__((ext_vector_type(16)));
typedef float  v8f   __attribute__((ext_vector_type(8)));

#define T_LEN  512
#define BATCH  128
#define DIN    400
#define HID    400
#define GATES  1600          // 4*H
#define KC0    13            // ceil(400/32): layer-0 input K-chunks (pad 416)
#define KCH    13            // recurrent K-chunks (H padded to 416)
#define KC1    25            // layer-1 input K-chunks (800 == 25*32, no pad)
#define MTILES 8             // 128/16 batch tiles
#define JT     25            // 400/16 hidden-column tiles
#define NTILES 100           // 1600/16 gate-column tiles
#define NWG    50            // block = (dir, jt); 8 waves = 8 batch tiles

// LDS: 128 lengths + worst-case weight slice 4*(KC1+KCH)*512 bf16 = 152KB
#define SMEM_BYTES (512 + 4 * (KC1 + KCH) * 512 * 2)

// -------- WMMA per-lane fragment index maps (CDNA5 ISA §7.12.2) -------
__device__ __forceinline__ int a_kk(int lane, int e) {
    return e + (e < 8 ? 0 : 8) + (lane >= 16 ? 8 : 0);
}
__device__ __forceinline__ int b_kk(int lane, int e) {
    return e + (e < 8 ? 0 : 8) + (lane >= 16 ? 8 : 0);
}

// Scatter one bf16 into a packed-A block ([KC][32 lanes][16 elems]) at row m,
// K index k. Inverse of a_kk.
__device__ __forceinline__ void store_apack(bf16_t* base, int k, int rowm, bf16_t val) {
    int kc = k >> 5, kk = k & 31;
    int lh, e;
    if      (kk <  8) { lh = 0; e = kk;      }
    else if (kk < 16) { lh = 1; e = kk - 8;  }
    else if (kk < 24) { lh = 0; e = kk - 8;  }
    else              { lh = 1; e = kk - 16; }
    base[((long)kc * 32 + lh * 16 + rowm) * 16 + e] = val;
}

__device__ __forceinline__ float fsigmoid(float x) {
    return 1.0f / (1.0f + __expf(-x));
}

// ------------------------------ grid barrier --------------------------------
__device__ __forceinline__ void grid_barrier(unsigned* cnt, unsigned* gen) {
    __threadfence();
    __syncthreads();
    if (threadIdx.x == 0) {
        unsigned g = __hip_atomic_load(gen, __ATOMIC_ACQUIRE, __HIP_MEMORY_SCOPE_AGENT);
        unsigned arrived = __hip_atomic_fetch_add(cnt, 1u, __ATOMIC_ACQ_REL, __HIP_MEMORY_SCOPE_AGENT);
        if (arrived == gridDim.x - 1) {
            __hip_atomic_store(cnt, 0u, __ATOMIC_RELAXED, __HIP_MEMORY_SCOPE_AGENT);
            __hip_atomic_fetch_add(gen, 1u, __ATOMIC_RELEASE, __HIP_MEMORY_SCOPE_AGENT);
        } else {
            while (__hip_atomic_load(gen, __ATOMIC_ACQUIRE, __HIP_MEMORY_SCOPE_AGENT) == g)
                __builtin_amdgcn_s_sleep(2);
        }
    }
    __syncthreads();
}

// ------------------------------ pack kernels --------------------------------
__global__ void zero_kernel(unsigned* __restrict__ p, long n) {
    for (long i = blockIdx.x * (long)blockDim.x + threadIdx.x; i < n;
         i += (long)gridDim.x * blockDim.x) p[i] = 0u;
}

// x [T,B,DIN] fp32 -> packed A bf16 [T][MTILES][KC0][32][16], K padded w/ zeros
__global__ void pack_x_kernel(const float* __restrict__ x, bf16_t* __restrict__ xp, long n) {
    for (long idx = blockIdx.x * (long)blockDim.x + threadIdx.x; idx < n;
         idx += (long)gridDim.x * blockDim.x) {
        int e    = idx & 15;
        int lane = (idx >> 4) & 31;
        long rest = idx >> 9;
        int kc = (int)(rest % KC0); rest /= KC0;
        int mt = (int)(rest % MTILES);
        long t = rest / MTILES;
        int b = mt * 16 + (lane & 15);
        int k = kc * 32 + a_kk(lane, e);
        float v = (k < DIN) ? x[(t * BATCH + b) * DIN + k] : 0.0f;
        xp[idx] = (bf16_t)v;
    }
}

// w [GATES,Kreal] fp32 (row-major) -> packed B bf16 [NTILES][KC][32][16] of w^T
__global__ void pack_w_kernel(const float* __restrict__ w, bf16_t* __restrict__ wp,
                              int Kreal, int KC, long n) {
    for (long idx = blockIdx.x * (long)blockDim.x + threadIdx.x; idx < n;
         idx += (long)gridDim.x * blockDim.x) {
        int e    = idx & 15;
        int lane = (idx >> 4) & 31;
        long rest = idx >> 9;
        int kc = (int)(rest % KC);
        int nt = (int)(rest / KC);
        int ncol = nt * 16 + (lane & 15);
        int k    = kc * 32 + b_kk(lane, e);
        float v = (k < Kreal) ? w[(long)ncol * Kreal + k] : 0.0f;
        wp[idx] = (bf16_t)v;
    }
}

__global__ void bias_kernel(const float* bi0f, const float* bh0f,
                            const float* bi0b, const float* bh0b,
                            const float* bi1f, const float* bh1f,
                            const float* bi1b, const float* bh1b,
                            float* __restrict__ bsum) {
    int idx = blockIdx.x * blockDim.x + threadIdx.x;
    if (idx >= 4 * GATES) return;
    int dl = idx / GATES, nn = idx % GATES;
    const float* a; const float* c;
    switch (dl) {
        case 0:  a = bi0f; c = bh0f; break;
        case 1:  a = bi0b; c = bh0b; break;
        case 2:  a = bi1f; c = bh1f; break;
        default: a = bi1b; c = bh1b; break;
    }
    bsum[idx] = a[nn] + c[nn];
}

// --------------------------- persistent LSTM kernel -------------------------
__global__ void __launch_bounds__(256, 1)
lstm_kernel(const bf16_t* __restrict__ xp0, bf16_t* __restrict__ xp1,
            const bf16_t* __restrict__ wih0, const bf16_t* __restrict__ whh0,
            const bf16_t* __restrict__ wih1, const bf16_t* __restrict__ whh1,
            const float*  __restrict__ bsum, const int* __restrict__ lengths,
            bf16_t* __restrict__ hpack,
            float* __restrict__ out, unsigned* bar_cnt, unsigned* bar_gen)
{
    extern __shared__ char smem_raw[];
    int*    s_len = (int*)smem_raw;
    bf16_t* s_w   = (bf16_t*)(smem_raw + 512);   // weight slice for this block

    for (int i = threadIdx.x; i < BATCH; i += blockDim.x) s_len[i] = lengths[i];

    const int lane = threadIdx.x & 31;
    const int dir  = blockIdx.x / JT;            // 0 = forward, 1 = backward
    const int jt   = blockIdx.x % JT;            // hidden-col tile (block-wide)
    const int mt   = threadIdx.x >> 5;           // batch tile = wave id 0..7
    const int jcol = jt * 16 + (lane & 15);
    const int rowm = (lane >= 16) ? 8 : 0;       // row base of D elements

    const long HP_PAR = (long)MTILES * KCH * 512;           // one parity buffer
    bf16_t* hp_dir = hpack + (long)dir * 2 * HP_PAR;

    for (int layer = 0; layer < 2; ++layer) {
        const bf16_t* xp  = layer ? xp1 : xp0;
        const int     KCI = layer ? KC1 : KC0;
        const bf16_t* wih = (layer ? wih1 : wih0) + (long)dir * NTILES * KCI * 512;
        const bf16_t* whh = (layer ? whh1 : whh0) + (long)dir * NTILES * KCH * 512;
        const float*  bs  = bsum + (long)(layer * 2 + dir) * GATES;

        // ---- stage this block's step-invariant weight slices into LDS ----
        bf16_t* s_wih = s_w;                      // [4][KCI][512]
        bf16_t* s_whh = s_w + 4 * KCI * 512;      // [4][KCH][512]
        for (int g = 0; g < 4; ++g) {
            const bf16_t* srcI = wih + (long)(g * JT + jt) * KCI * 512;
            bf16_t*       dstI = s_wih + g * KCI * 512;
            for (int i = threadIdx.x * 16; i < KCI * 512; i += blockDim.x * 16)
                *(v16bf*)(dstI + i) = *(const v16bf*)(srcI + i);
            const bf16_t* srcH = whh + (long)(g * JT + jt) * KCH * 512;
            bf16_t*       dstH = s_whh + g * KCH * 512;
            for (int i = threadIdx.x * 16; i < KCH * 512; i += blockDim.x * 16)
                *(v16bf*)(dstH + i) = *(const v16bf*)(srcH + i);
        }
        __syncthreads();

        // ---- reset this task's h slices (both parities); c lives in regs ----
        for (int par = 0; par < 2; ++par) {
            bf16_t* hb = hp_dir + (long)par * HP_PAR + (long)mt * KCH * 512;
            #pragma unroll
            for (int v = 0; v < 8; ++v)
                store_apack(hb, jcol, v + rowm, (bf16_t)0.0f);
        }
        float creg[8];
        #pragma unroll
        for (int v = 0; v < 8; ++v) creg[v] = 0.0f;
        grid_barrier(bar_cnt, bar_gen);

        float bias[4];
        #pragma unroll
        for (int g = 0; g < 4; ++g) bias[g] = bs[g * HID + jcol];

        for (int s = 0; s < T_LEN; ++s) {
            const int t    = dir ? (T_LEN - 1 - s) : s;
            const int rpar = s & 1;
            const int wpar = rpar ^ 1;

            v8f acc[4];
            #pragma unroll
            for (int g = 0; g < 4; ++g) {
                #pragma unroll
                for (int q = 0; q < 8; ++q) acc[g][q] = bias[g];
            }

            // ---- input contribution: x_t @ W_ih^T (A global, B from LDS) ----
            const bf16_t* abase = xp + ((long)t * MTILES + mt) * KCI * 512;
            {   // prefetch next timestep's A fragments into cache
                int tn = dir ? (t - 1) : (t + 1);
                if (tn >= 0 && tn < T_LEN)
                    __builtin_prefetch(xp + ((long)tn * MTILES + mt) * KCI * 512 + lane * 16, 0, 1);
            }
            v16bf a = *(const v16bf*)(abase + lane * 16);
            for (int kc = 0; kc < KCI; ++kc) {
                int kcn = (kc + 1 < KCI) ? kc + 1 : kc;      // pipelined A load
                v16bf an = *(const v16bf*)(abase + (long)kcn * 512 + lane * 16);
                #pragma unroll
                for (int g = 0; g < 4; ++g) {
                    v16bf bm = *(const v16bf*)(s_wih + (g * KCI + kc) * 512 + lane * 16);
                    acc[g] = __builtin_amdgcn_wmma_f32_16x16x32_bf16(
                        false, a, false, bm, (short)0, acc[g], false, false);
                }
                a = an;
            }
            // ---- recurrent contribution: h_{t-1} @ W_hh^T ----
            const bf16_t* hbase = hp_dir + (long)rpar * HP_PAR + (long)mt * KCH * 512;
            a = *(const v16bf*)(hbase + lane * 16);
            for (int kc = 0; kc < KCH; ++kc) {
                int kcn = (kc + 1 < KCH) ? kc + 1 : kc;
                v16bf an = *(const v16bf*)(hbase + (long)kcn * 512 + lane * 16);
                #pragma unroll
                for (int g = 0; g < 4; ++g) {
                    v16bf bm = *(const v16bf*)(s_whh + (g * KCH + kc) * 512 + lane * 16);
                    acc[g] = __builtin_amdgcn_wmma_f32_16x16x32_bf16(
                        false, a, false, bm, (short)0, acc[g], false, false);
                }
                a = an;
            }

            // ---- LSTM cell + masking + stores (c stays in registers) ----
            bf16_t* hw = hp_dir + (long)wpar * HP_PAR + (long)mt * KCH * 512;
            bf16_t* xw = layer ? (bf16_t*)0
                               : xp1 + ((long)t * MTILES + mt) * KC1 * 512;
            #pragma unroll
            for (int v = 0; v < 8; ++v) {
                int rm = v + rowm;
                int b  = mt * 16 + rm;
                float iv = fsigmoid(acc[0][v]);
                float fv = fsigmoid(acc[1][v]);
                float gv = tanhf(acc[2][v]);
                float ov = fsigmoid(acc[3][v]);
                float cn = fv * creg[v] + iv * gv;
                float hn = ov * tanhf(cn);
                if (t >= s_len[b]) { cn = 0.0f; hn = 0.0f; }
                creg[v] = cn;
                store_apack(hw, jcol, rm, (bf16_t)hn);
                if (layer == 0) {
                    store_apack(xw, dir * HID + jcol, rm, (bf16_t)hn);
                } else {
                    out[((long)t * BATCH + b) * (2 * HID) + dir * HID + jcol] = hn;
                }
            }
            grid_barrier(bar_cnt, bar_gen);
        }
    }
}

// ------------------------------- launcher -----------------------------------
extern "C" void kernel_launch(void* const* d_in, const int* in_sizes, int n_in,
                              void* d_out, int out_size, void* d_ws, size_t ws_size,
                              hipStream_t stream) {
    (void)in_sizes; (void)n_in; (void)out_size; (void)ws_size;

    const float* x       = (const float*)d_in[0];
    const int*   lengths = (const int*)  d_in[1];
    const float* w_ih_f0 = (const float*)d_in[2];
    const float* w_hh_f0 = (const float*)d_in[3];
    const float* b_ih_f0 = (const float*)d_in[4];
    const float* b_hh_f0 = (const float*)d_in[5];
    const float* w_ih_b0 = (const float*)d_in[6];
    const float* w_hh_b0 = (const float*)d_in[7];
    const float* b_ih_b0 = (const float*)d_in[8];
    const float* b_hh_b0 = (const float*)d_in[9];
    const float* w_ih_f1 = (const float*)d_in[10];
    const float* w_hh_f1 = (const float*)d_in[11];
    const float* b_ih_f1 = (const float*)d_in[12];
    const float* b_hh_f1 = (const float*)d_in[13];
    const float* w_ih_b1 = (const float*)d_in[14];
    const float* w_hh_b1 = (const float*)d_in[15];
    const float* b_ih_b1 = (const float*)d_in[16];
    const float* b_hh_b1 = (const float*)d_in[17];
    float* out = (float*)d_out;

    char* ws = (char*)d_ws;
    size_t off = 0;
    auto alloc = [&](size_t bytes) -> char* {
        char* p = ws + off;
        off = (off + bytes + 255) & ~(size_t)255;
        return p;
    };

    unsigned* bar    = (unsigned*)alloc(256);                                  // [cnt, gen]
    float*    bsum   = (float*)   alloc((size_t)4 * GATES * 4);
    bf16_t*   hpack  = (bf16_t*)  alloc((size_t)2 * 2 * MTILES * KCH * 512 * 2);
    bf16_t*   wih0   = (bf16_t*)  alloc((size_t)2 * NTILES * KC0 * 512 * 2);
    bf16_t*   whh0   = (bf16_t*)  alloc((size_t)2 * NTILES * KCH * 512 * 2);
    bf16_t*   wih1   = (bf16_t*)  alloc((size_t)2 * NTILES * KC1 * 512 * 2);
    bf16_t*   whh1   = (bf16_t*)  alloc((size_t)2 * NTILES * KCH * 512 * 2);
    bf16_t*   xp0    = (bf16_t*)  alloc((size_t)T_LEN * MTILES * KC0 * 512 * 2);
    bf16_t*   xp1    = (bf16_t*)  alloc((size_t)T_LEN * MTILES * KC1 * 512 * 2);

    // ---- per-launch state reset (graph-replay deterministic) ----
    zero_kernel<<<1, 64, 0, stream>>>(bar, 64);
    {
        long n = ((long)2 * 2 * MTILES * KCH * 512 * 2) / 4;  // hpack as u32
        zero_kernel<<<(int)((n + 255) / 256), 256, 0, stream>>>((unsigned*)hpack, n);
    }

    // ---- pack activations & weights into WMMA fragment layouts ----
    {
        long n = (long)T_LEN * MTILES * KC0 * 512;
        pack_x_kernel<<<8192, 256, 0, stream>>>(x, xp0, n);
    }
    const long nW13 = (long)NTILES * KC0 * 512;  // 100*13*512
    const long nW25 = (long)NTILES * KC1 * 512;  // 100*25*512
    int g13 = (int)((nW13 + 255) / 256), g25 = (int)((nW25 + 255) / 256);
    pack_w_kernel<<<g13, 256, 0, stream>>>(w_ih_f0, wih0,            DIN, KC0, nW13);
    pack_w_kernel<<<g13, 256, 0, stream>>>(w_ih_b0, wih0 + nW13,     DIN, KC0, nW13);
    pack_w_kernel<<<g13, 256, 0, stream>>>(w_hh_f0, whh0,            HID, KCH, nW13);
    pack_w_kernel<<<g13, 256, 0, stream>>>(w_hh_b0, whh0 + nW13,     HID, KCH, nW13);
    pack_w_kernel<<<g25, 256, 0, stream>>>(w_ih_f1, wih1,        2 * HID, KC1, nW25);
    pack_w_kernel<<<g25, 256, 0, stream>>>(w_ih_b1, wih1 + nW25, 2 * HID, KC1, nW25);
    pack_w_kernel<<<g13, 256, 0, stream>>>(w_hh_f1, whh1,            HID, KCH, nW13);
    pack_w_kernel<<<g13, 256, 0, stream>>>(w_hh_b1, whh1 + nW13,     HID, KCH, nW13);
    bias_kernel<<<(4 * GATES + 255) / 256, 256, 0, stream>>>(
        b_ih_f0, b_hh_f0, b_ih_b0, b_hh_b0, b_ih_f1, b_hh_f1, b_ih_b1, b_hh_b1, bsum);

    // ---- persistent bidirectional LSTM ----
    lstm_kernel<<<NWG, 256, SMEM_BYTES, stream>>>(xp0, xp1, wih0, whh0, wih1, whh1,
                                                  bsum, lengths, hpack,
                                                  out, bar, bar + 1);
}
